// NERModel_14250701488262
// MI455X (gfx1250) — matrix operand order repaired and checked by reading, not compiled, not run on hardware
//
#include <hip/hip_runtime.h>
#include <cstddef>

typedef __attribute__((ext_vector_type(16))) _Float16 v16h;
typedef __attribute__((ext_vector_type(8)))  float    v8f;
typedef __attribute__((ext_vector_type(4)))  unsigned int v4u;
typedef __attribute__((ext_vector_type(8)))  int      v8i32;
typedef __attribute__((ext_vector_type(4)))  int      v4i32;

#define TAGS   12
#define TSTART 10
#define TSTOP  11
#define DWRD   300
#define DCH    25
#define HCH    25
#define HWD    256
#define BB     64
#define LL     128
#define WW     16
#define KX     352          // 350 padded to 11 K-tiles of 32
#define GW     1024         // 4*HWD
#define PP     (LL*BB)      // 8192 sequence positions

__device__ __forceinline__ float sigf(float x) { return 1.0f / (1.0f + __expf(-x)); }

// ---------------------------------------------------------------------------
// TDM: async-load a 64 x 1024 f16 tile (row stride 1024) from global into LDS.
// D# per CDNA5 ISA ch.8: group0 = {count/type, lds_addr, global_addr},
// group1 = {data_size, tensor dims, tile dims, stride}; groups 2/3/4 zero (2D).
// Tracked by TENSORcnt. 6-arg builtin form (clang-23 / therock-10.0 headers).
// ---------------------------------------------------------------------------
__device__ __forceinline__ void tdm_load_gx_tile(const _Float16* gsrc, unsigned lds_off)
{
    unsigned long long ga = (unsigned long long)(size_t)gsrc;
    v4u g0;
    g0[0] = 1u;                                             // count=1 (valid user D#)
    g0[1] = lds_off;                                        // LDS byte address
    g0[2] = (unsigned)(ga & 0xFFFFFFFFu);                   // global_addr[31:0]
    g0[3] = (unsigned)((ga >> 32) & 0x01FFFFFFu)            // global_addr[56:32]
          | (2u << 30);                                     // type = 2 ("image")
    v8i32 g1;
    g1[0] = 0x00010000;                                     // data_size=1 -> 2 bytes
    g1[1] = (int)((1024u & 0xFFFFu) << 16);                 // tensor_dim0 = 1024 (lo16)
    g1[2] = (int)((64u & 0xFFFFu) << 16);                   // dim0 hi=0 | tensor_dim1=64
    g1[3] = (int)((1024u & 0xFFFFu) << 16);                 // dim1 hi=0 | tile_dim0=1024
    g1[4] = 64;                                             // tile_dim1=64, tile_dim2=0
    g1[5] = 1024;                                           // tensor_dim0_stride=1024
    g1[6] = 0;
    g1[7] = 0;
    v4i32 gz4 = {0, 0, 0, 0};
    v8i32 gz8 = {0, 0, 0, 0, 0, 0, 0, 0};
    __builtin_amdgcn_tensor_load_to_lds(g0, g1, gz4, gz4, gz8, 0);
}

// ---------------------------------------------------------------------------
// K0: convert word-LSTM weights to f16, n-major (B-operand friendly), pad K,
//     and fold biases (bih+bhh).
// ---------------------------------------------------------------------------
__global__ void k_prep_weights(const float* __restrict__ Wih_wf, const float* __restrict__ Whh_wf,
                               const float* __restrict__ bih_wf, const float* __restrict__ bhh_wf,
                               const float* __restrict__ Wih_wb, const float* __restrict__ Whh_wb,
                               const float* __restrict__ bih_wb, const float* __restrict__ bhh_wb,
                               _Float16* __restrict__ Wn, _Float16* __restrict__ Un,
                               float* __restrict__ bias)
{
    const int n1 = 2 * GW * KX;
    const int n2 = 2 * GW * HWD;
    const int n3 = 2 * GW;
    for (int idx = blockIdx.x * blockDim.x + threadIdx.x; idx < n1 + n2 + n3;
         idx += gridDim.x * blockDim.x) {
        if (idx < n1) {
            int d = idx / (GW * KX), rem = idx % (GW * KX);
            int n = rem / KX, k = rem % KX;
            const float* Wih = d ? Wih_wb : Wih_wf;
            Wn[idx] = (_Float16)(k < 350 ? Wih[n * 350 + k] : 0.0f);
        } else if (idx < n1 + n2) {
            int j = idx - n1;
            int d = j / (GW * HWD), rem = j % (GW * HWD);
            const float* Whh = d ? Whh_wb : Whh_wf;
            Un[j] = (_Float16)Whh[rem];
        } else {
            int j = idx - n1 - n2;
            int d = j >> 10, n = j & 1023;
            bias[j] = d ? (bih_wb[n] + bhh_wb[n]) : (bih_wf[n] + bhh_wf[n]);
        }
    }
}

// ---------------------------------------------------------------------------
// K1: char BiLSTM, one thread per word; weights + char embedding table in LDS,
//     h/c state fully unrolled into VGPRs. Writes cfeat (B*L, 50).
// ---------------------------------------------------------------------------
__global__ __launch_bounds__(256) void k_char_lstm(
    const int* __restrict__ chars, const float* __restrict__ char_emb,
    const float* __restrict__ Wih_cf, const float* __restrict__ Whh_cf,
    const float* __restrict__ bih_cf, const float* __restrict__ bhh_cf,
    const float* __restrict__ Wih_cb, const float* __restrict__ Whh_cb,
    const float* __restrict__ bih_cb, const float* __restrict__ bhh_cb,
    float* __restrict__ cfeat)
{
    __shared__ float s_emb[100 * DCH];
    __shared__ float s_Wih[2][100 * DCH];
    __shared__ float s_Whh[2][100 * DCH];
    __shared__ float s_b[2][100];
    for (int i = threadIdx.x; i < 100 * DCH; i += blockDim.x) {
        s_emb[i]    = char_emb[i];
        s_Wih[0][i] = Wih_cf[i];  s_Whh[0][i] = Whh_cf[i];
        s_Wih[1][i] = Wih_cb[i];  s_Whh[1][i] = Whh_cb[i];
    }
    for (int i = threadIdx.x; i < 100; i += blockDim.x) {
        s_b[0][i] = bih_cf[i] + bhh_cf[i];
        s_b[1][i] = bih_cb[i] + bhh_cb[i];
    }
    __syncthreads();

    int word = blockIdx.x * blockDim.x + threadIdx.x;   // = b*L + l
    if (word >= PP) return;

    float h[HCH], c[HCH], hn[HCH], x[DCH];
    for (int d = 0; d < 2; ++d) {
        #pragma unroll
        for (int j = 0; j < HCH; ++j) { h[j] = 0.0f; c[j] = 0.0f; }
        for (int s = 0; s < WW; ++s) {
            int w = d ? (WW - 1 - s) : s;
            int ci = chars[word * WW + w];
            #pragma unroll
            for (int k = 0; k < DCH; ++k) x[k] = s_emb[ci * DCH + k];
            #pragma unroll
            for (int j = 0; j < HCH; ++j) {
                float gi = s_b[d][j], gf = s_b[d][25 + j], gg = s_b[d][50 + j], go = s_b[d][75 + j];
                #pragma unroll
                for (int k = 0; k < DCH; ++k) {
                    gi += s_Wih[d][j * 25 + k]        * x[k] + s_Whh[d][j * 25 + k]        * h[k];
                    gf += s_Wih[d][(25 + j) * 25 + k] * x[k] + s_Whh[d][(25 + j) * 25 + k] * h[k];
                    gg += s_Wih[d][(50 + j) * 25 + k] * x[k] + s_Whh[d][(50 + j) * 25 + k] * h[k];
                    go += s_Wih[d][(75 + j) * 25 + k] * x[k] + s_Whh[d][(75 + j) * 25 + k] * h[k];
                }
                float cc = sigf(gf) * c[j] + sigf(gi) * tanhf(gg);
                c[j]  = cc;
                hn[j] = sigf(go) * tanhf(cc);
            }
            #pragma unroll
            for (int j = 0; j < HCH; ++j) h[j] = hn[j];
        }
        #pragma unroll
        for (int j = 0; j < HCH; ++j) cfeat[word * 50 + d * 25 + j] = h[j];
    }
}

// ---------------------------------------------------------------------------
// K2: build word-LSTM input X (time-major, f16, K padded to 352)
// ---------------------------------------------------------------------------
__global__ void k_build_x(const int* __restrict__ words, const float* __restrict__ word_emb,
                          const float* __restrict__ cfeat, _Float16* __restrict__ X)
{
    for (int idx = blockIdx.x * blockDim.x + threadIdx.x; idx < PP * KX;
         idx += gridDim.x * blockDim.x) {
        int p = idx / KX, k = idx % KX;
        int t = p / BB, b = p % BB;
        float v = 0.0f;
        if (k < DWRD)            v = word_emb[(size_t)words[b * LL + t] * DWRD + k];
        else if (k < DWRD + 50)  v = cfeat[(b * LL + t) * 50 + (k - DWRD)];
        X[idx] = (_Float16)v;
    }
}

// ---------------------------------------------------------------------------
// K3: big WMMA GEMM, both directions: gx[d] = X(8192x352) * Wn[d]ᵀ + bias[d].
// ---------------------------------------------------------------------------
__global__ __launch_bounds__(256) void k_gemm_gx(const _Float16* __restrict__ X,
                                                 const _Float16* __restrict__ Wn,
                                                 const float* __restrict__ bias,
                                                 _Float16* __restrict__ gx)
{
    int dir   = blockIdx.x >> 9;
    int mtile = blockIdx.x & 511;
    const _Float16* W  = Wn   + (size_t)dir * GW * KX;
    const float*    bs = bias + (size_t)dir * GW;
    _Float16*       G  = gx   + (size_t)dir * PP * GW;

    int wave = threadIdx.x >> 5;
    int lane = threadIdx.x & 31;
    int half = lane >> 4;
    int lane15 = lane & 15;
    int ntbase = wave * 8;

    v8f acc[8];
    #pragma unroll
    for (int i = 0; i < 8; ++i) acc[i] = (v8f){0, 0, 0, 0, 0, 0, 0, 0};

    const _Float16* Xrow = X + (size_t)(mtile * 16 + lane15) * KX;

    for (int kt = 0; kt < 11; ++kt) {
        v16h a;
        #pragma unroll
        for (int r = 0; r < 8; ++r) {                     // A: 16x32 f16 layout
            int k = kt * 32 + ((r < 4) ? 2 * r : 2 * r + 8) + 8 * half;
            a[2 * r]     = Xrow[k];
            a[2 * r + 1] = Xrow[k + 1];
        }
        #pragma unroll
        for (int i = 0; i < 8; ++i) {
            int nt = ntbase + i;
            const _Float16* Wrow = W + (size_t)(nt * 16 + lane15) * KX + kt * 32 + half * 8;
            v16h bm;
            #pragma unroll
            for (int r = 0; r < 8; ++r) {                 // B: 32x16, same K order as A
                int kk = (r < 4) ? 2 * r : 2 * r + 8;
                bm[2 * r]     = Wrow[kk];
                bm[2 * r + 1] = Wrow[kk + 1];
            }
            acc[i] = __builtin_amdgcn_wmma_f32_16x16x32_f16(
                false, a, false, bm, (short)0, acc[i], false, false);
        }
    }
    #pragma unroll
    for (int i = 0; i < 8; ++i) {
        int n = (ntbase + i) * 16 + lane15;
        float bb = bs[n];
        #pragma unroll
        for (int r = 0; r < 8; ++r) {                     // D: M = r + 8*half, N = lane15
            int p = mtile * 16 + r + 8 * half;
            G[(size_t)p * GW + n] = (_Float16)(acc[i][r] + bb);
        }
    }
}

// ---------------------------------------------------------------------------
// K4: recurrent word LSTM. 2 persistent blocks (fwd/bwd), 1024 threads each.
//     gx[t] tile (64x1024 f16, 128KB) streamed into LDS by the Tensor Data
//     Mover one step ahead (TENSORcnt / s_wait_tensorcnt), overlapping the
//     gate nonlinearity and h update of the current step.
//     LDS: 32KB H(f16) + 128KB G(f16) + 128KB gx stage = 288KB (<320KB/WGP).
// ---------------------------------------------------------------------------
__global__ __launch_bounds__(1024) void k_word_lstm(const _Float16* __restrict__ gx,
                                                    const _Float16* __restrict__ Un,
                                                    float* __restrict__ hs)
{
    extern __shared__ char smem[];
    _Float16* Hh    = (_Float16*)smem;                          // 64*256  f16 = 32KB
    _Float16* Gsh   = (_Float16*)(smem + BB * HWD * 2);         // 64*1024 f16 = 128KB
    _Float16* stage = (_Float16*)(smem + BB * HWD * 2 + BB * GW * 2); // 128KB
    unsigned stage_off = (unsigned)(size_t)stage;               // LDS byte offset (addr[31:0])

    int dir = blockIdx.x;
    const _Float16* GX = gx + (size_t)dir * PP * GW;
    const _Float16* U  = Un + (size_t)dir * GW * HWD;

    int tid  = threadIdx.x;
    int wave = tid >> 5, lane = tid & 31;
    int half = lane >> 4, lane15 = lane & 15;
    int mtile  = wave >> 3;                               // 0..3  (16 batch rows each)
    int ntbase = (wave & 7) * 8;                          // 8 N-tiles per wave

    int eb = tid >> 4;                                    // batch row owned (elementwise)
    int ej = (tid & 15) * 16;                             // 16 hidden units owned

    float cst[16];
    #pragma unroll
    for (int q = 0; q < 16; ++q) cst[q] = 0.0f;
    for (int i = tid; i < BB * HWD; i += 1024) Hh[i] = (_Float16)0.0f;

    // pre-issue TDM for the first step's gx tile
    if (tid < 32) {
        int t0 = dir ? (LL - 1) : 0;
        tdm_load_gx_tile(GX + (size_t)(t0 * BB) * GW, stage_off);
    }
    __syncthreads();

    for (int s = 0; s < LL; ++s) {
        int t = dir ? (LL - 1 - s) : s;

        if (tid < 32) __builtin_amdgcn_s_wait_tensorcnt(0);   // stage tile landed
        __syncthreads();

        #pragma unroll
        for (int i = 0; i < 8; ++i) {
            int n = (ntbase + i) * 16 + lane15;
            v8f acc;
            #pragma unroll
            for (int r = 0; r < 8; ++r)
                acc[r] = (float)stage[(mtile * 16 + r + 8 * half) * GW + n];
            const _Float16* Hrow = Hh + (mtile * 16 + lane15) * HWD;
            #pragma unroll
            for (int kt = 0; kt < 8; ++kt) {
                v16h a, bm;
                #pragma unroll
                for (int r = 0; r < 8; ++r) {
                    int k = kt * 32 + ((r < 4) ? 2 * r : 2 * r + 8) + 8 * half;
                    a[2 * r]     = Hrow[k];
                    a[2 * r + 1] = Hrow[k + 1];
                }
                const _Float16* Urow = U + (size_t)n * HWD + kt * 32 + half * 8;
                #pragma unroll
                for (int r = 0; r < 8; ++r) {
                    int kk = (r < 4) ? 2 * r : 2 * r + 8;
                    bm[2 * r]     = Urow[kk];
                    bm[2 * r + 1] = Urow[kk + 1];
                }
                acc = __builtin_amdgcn_wmma_f32_16x16x32_f16(
                    false, a, false, bm, (short)0, acc, false, false);
            }
            #pragma unroll
            for (int r = 0; r < 8; ++r)
                Gsh[(mtile * 16 + r + 8 * half) * GW + n] = (_Float16)acc[r];
        }
        __syncthreads();   // stage reads + Gsh writes complete

        // overlap: DMA next step's gx tile while we do the nonlinearity
        if (s + 1 < LL && tid < 32) {
            int tn = dir ? (LL - 2 - s) : (s + 1);
            tdm_load_gx_tile(GX + (size_t)(tn * BB) * GW, stage_off);
        }

        float hv[16];
        #pragma unroll
        for (int q = 0; q < 16; ++q) {
            int j = ej + q;
            float gi = (float)Gsh[eb * GW + j];
            float gf = (float)Gsh[eb * GW + 256 + j];
            float gg = (float)Gsh[eb * GW + 512 + j];
            float go = (float)Gsh[eb * GW + 768 + j];
            float cc = sigf(gf) * cst[q] + sigf(gi) * tanhf(gg);
            cst[q] = cc;
            hv[q]  = sigf(go) * tanhf(cc);
        }
        __syncthreads();
        #pragma unroll
        for (int q = 0; q < 16; ++q) {
            int j = ej + q;
            Hh[eb * HWD + j] = (_Float16)hv[q];
            hs[((size_t)t * BB + eb) * (2 * HWD) + dir * HWD + j] = hv[q];
        }
        __syncthreads();
    }
}

// ---------------------------------------------------------------------------
// K5: emission projection feats(L,B,12) = hs(·,512) @ Woᵀ + bo
// ---------------------------------------------------------------------------
__global__ void k_proj(const float* __restrict__ hs, const float* __restrict__ Wo,
                       const float* __restrict__ bo, float* __restrict__ feats)
{
    for (int idx = blockIdx.x * blockDim.x + threadIdx.x; idx < PP * TAGS;
         idx += gridDim.x * blockDim.x) {
        int p = idx / TAGS, tag = idx % TAGS;
        float sum = bo[tag];
        const float* hrow = hs + (size_t)p * (2 * HWD);
        const float* wrow = Wo + (size_t)tag * (2 * HWD);
        for (int k = 0; k < 2 * HWD; ++k) sum += hrow[k] * wrow[k];
        feats[idx] = sum;
    }
}

// ---------------------------------------------------------------------------
// K6: CRF forward (log Z). Single block, alpha in LDS, 128 sequential steps.
// ---------------------------------------------------------------------------
__global__ __launch_bounds__(256) void k_crf(const float* __restrict__ feats,
                                             const float* __restrict__ trans,
                                             float* __restrict__ out)
{
    __shared__ float alpha[BB * TAGS], alphan[BB * TAGS], ft[BB * TAGS], tr[TAGS * TAGS];
    int tid = threadIdx.x;
    for (int i = tid; i < TAGS * TAGS; i += 256) tr[i] = trans[i];
    for (int i = tid; i < BB * TAGS; i += 256) alpha[i] = ((i % TAGS) == TSTART) ? 0.0f : -10000.0f;
    __syncthreads();
    for (int t = 0; t < LL; ++t) {
        for (int i = tid; i < BB * TAGS; i += 256) ft[i] = feats[t * BB * TAGS + i];
        __syncthreads();
        for (int i = tid; i < BB * TAGS; i += 256) {
            int b = i / TAGS, nx = i % TAGS;
            float m = -3.0e38f;
            #pragma unroll
            for (int p = 0; p < TAGS; ++p) m = fmaxf(m, alpha[b * TAGS + p] + tr[nx * TAGS + p]);
            float ssum = 0.0f;
            #pragma unroll
            for (int p = 0; p < TAGS; ++p) ssum += __expf(alpha[b * TAGS + p] + tr[nx * TAGS + p] - m);
            alphan[i] = m + __logf(ssum) + ft[i];
        }
        __syncthreads();
        for (int i = tid; i < BB * TAGS; i += 256) alpha[i] = alphan[i];
        __syncthreads();
    }
    if (tid < BB) {
        float m = -3.0e38f;
        #pragma unroll
        for (int p = 0; p < TAGS; ++p) m = fmaxf(m, alpha[tid * TAGS + p] + tr[TSTOP * TAGS + p]);
        float ssum = 0.0f;
        #pragma unroll
        for (int p = 0; p < TAGS; ++p) ssum += __expf(alpha[tid * TAGS + p] + tr[TSTOP * TAGS + p] - m);
        out[tid] = m + __logf(ssum);
    }
}

// ---------------------------------------------------------------------------
extern "C" void kernel_launch(void* const* d_in, const int* in_sizes, int n_in,
                              void* d_out, int out_size, void* d_ws, size_t ws_size,
                              hipStream_t stream)
{
    const int*   words    = (const int*)  d_in[0];
    const int*   chars    = (const int*)  d_in[1];
    const float* word_emb = (const float*)d_in[2];
    const float* char_emb = (const float*)d_in[3];
    const float* Wih_cf = (const float*)d_in[4],  *Whh_cf = (const float*)d_in[5];
    const float* bih_cf = (const float*)d_in[6],  *bhh_cf = (const float*)d_in[7];
    const float* Wih_cb = (const float*)d_in[8],  *Whh_cb = (const float*)d_in[9];
    const float* bih_cb = (const float*)d_in[10], *bhh_cb = (const float*)d_in[11];
    const float* Wih_wf = (const float*)d_in[12], *Whh_wf = (const float*)d_in[13];
    const float* bih_wf = (const float*)d_in[14], *bhh_wf = (const float*)d_in[15];
    const float* Wih_wb = (const float*)d_in[16], *Whh_wb = (const float*)d_in[17];
    const float* bih_wb = (const float*)d_in[18], *bhh_wb = (const float*)d_in[19];
    const float* Wo     = (const float*)d_in[20], *bo     = (const float*)d_in[21];
    const float* trans  = (const float*)d_in[22];
    float* out = (float*)d_out;

    char* ws = (char*)d_ws;
    size_t off = 0;
    auto carve = [&](size_t bytes) -> void* {
        void* p = ws + off;
        off = (off + bytes + 255) & ~(size_t)255;
        return p;
    };
    float*    cfeat = (float*)   carve((size_t)PP * 50 * 4);
    _Float16* X     = (_Float16*)carve((size_t)PP * KX * 2);
    _Float16* Wn    = (_Float16*)carve((size_t)2 * GW * KX * 2);
    _Float16* Un    = (_Float16*)carve((size_t)2 * GW * HWD * 2);
    float*    bias  = (float*)   carve((size_t)2 * GW * 4);
    _Float16* gx    = (_Float16*)carve((size_t)2 * PP * GW * 2);
    float*    hsbuf = (float*)   carve((size_t)PP * 2 * HWD * 4);
    float*    feats = (float*)   carve((size_t)PP * TAGS * 4);

    k_prep_weights<<<2048, 256, 0, stream>>>(Wih_wf, Whh_wf, bih_wf, bhh_wf,
                                             Wih_wb, Whh_wb, bih_wb, bhh_wb,
                                             Wn, Un, bias);
    k_char_lstm<<<PP / 256, 256, 0, stream>>>(chars, char_emb,
                                              Wih_cf, Whh_cf, bih_cf, bhh_cf,
                                              Wih_cb, Whh_cb, bih_cb, bhh_cb, cfeat);
    k_build_x<<<2048, 256, 0, stream>>>(words, word_emb, cfeat, X);
    k_gemm_gx<<<1024, 256, 0, stream>>>(X, Wn, bias, gx);
    // LDS: 32KB H + 128KB G + 128KB gx stage = 294912 B (< 320KB/WGP)
    k_word_lstm<<<2, 1024, BB * HWD * 2 + BB * GW * 2 + BB * GW * 2, stream>>>(gx, Un, hsbuf);
    k_proj<<<384, 256, 0, stream>>>(hsbuf, Wo, bo, feats);
    k_crf<<<1, 256, 0, stream>>>(feats, trans, out);
}
